// SpatialAwareAttention_56959856279791
// MI455X (gfx1250) — matrix operand order repaired and checked by reading, not compiled, not run on hardware
//
#include <hip/hip_runtime.h>
#include <hip/hip_bf16.h>
#include <math.h>

// MI455X / gfx1250, wave32. WMMA f32_16x16x32_f16 implicit GEMMs for both convs.
// x is staged as zero-padded NHWC f16 so the GEMM inner loops are branchless.

typedef __attribute__((ext_vector_type(16))) _Float16 v16h;
typedef __attribute__((ext_vector_type(8)))  float    v8f;

#define BATCH 8
#define CCH   256
#define HWPX  4096     // 64*64
#define PADW  66
#define PADA  4356     // 66*66
#define GRP   4
#define CG    64
#define KKTAP 9

// K index inside a 32-wide chunk in the contiguous order this code uses for both WMMA A and B:
// lanes 0-15: M=lane, elements 0..15 hold K={0..15}
// lanes16-31: M=lane-16, elements 0..15 hold K={16..31}
__device__ __forceinline__ int a_layout_k(int lane, int e) {
  return (lane & 16) + e;
}

// ---- Stage 0: x (NCHW f32) -> xh (padded NHWC f16, zero halo) ----------------
// grid (66*66, 8), block 256: row/col/b are block-uniform (SALU), thread = channel
__global__ void k_convert_x(const float* __restrict__ x, _Float16* __restrict__ xh) {
  int row = blockIdx.x / PADW;        // uniform scalar math
  int col = blockIdx.x % PADW;
  int b = blockIdx.y;
  int c = threadIdx.x;
  float v = 0.f;
  if (row >= 1 && row <= 64 && col >= 1 && col <= 64)
    v = x[(((size_t)b * CCH + c) << 12) + ((row - 1) << 6) + (col - 1)];
  xh[((size_t)b * PADA + blockIdx.x) * CCH + c] = (_Float16)v;
}

// ---- Stage 0b: pack offset-conv weights into WMMA-A layout -------------------
// packed: [7 m-tiles][72 k-chunks][32 lanes][16 f16]   (K order: k = kk*256 + c)
// grid (72, 7), block 512 = 32 lanes x 16 elems
__global__ void k_pack_woff(const float* __restrict__ w, _Float16* __restrict__ pk) {
  int e = threadIdx.x & 15;
  int lane = threadIdx.x >> 4;
  int ch = blockIdx.x;
  int mt = blockIdx.y;
  int m = mt * 16 + (lane & 15);
  int k = ch * 32 + a_layout_k(lane, e);
  int kk = k >> 8, c = k & 255;
  float v = (m < 108) ? w[(m * CCH + c) * KKTAP + kk] : 0.f;
  pk[(size_t)(mt * 72 + ch) * 512 + threadIdx.x] = (_Float16)v;
}

// ---- Stage 0c: pack grouped dcn weights --------------------------------------
// packed: [G][4 m-tiles][18 k-chunks][32 lanes][16 f16]  (K order: k = kk*64 + c)
// grid (18, 16), block 512
__global__ void k_pack_wdcn(const float* __restrict__ w, _Float16* __restrict__ pk) {
  int e = threadIdx.x & 15;
  int lane = threadIdx.x >> 4;
  int ch = blockIdx.x;
  int gt = blockIdx.y;                // g*4 + t
  int t = gt & 3, g = gt >> 2;
  int oc = t * 16 + (lane & 15);
  int k = ch * 32 + a_layout_k(lane, e);
  int kk = k >> 6, c = k & 63;
  pk[(size_t)(gt * 18 + ch) * 512 + threadIdx.x] = (_Float16)w[((g * CG + oc) * CG + c) * KKTAP + kk];
}

// ---- Stage 1: offset/mask conv as implicit GEMM (M=108, K=2304, N=32768) -----
// one wave per 16-pixel tile; 7 accumulators (all M tiles) reuse each B chunk.
// Padded input => branchless inner loop: 1 B load + 7 A loads + 7 WMMAs.
__global__ void __launch_bounds__(128)
k_conv_off(const _Float16* __restrict__ xh, const _Float16* __restrict__ wpk,
           const float* __restrict__ bias, float* __restrict__ off) {
  int lane = threadIdx.x & 31;
  int wv = threadIdx.x >> 5;
  int lo = lane & 15;
  int hi = (lane >> 4) & 1;
  int tile = blockIdx.x * 4 + wv;     // 0..2047 : b*256 + pixel-tile
  int b = tile >> 8;
  int n0 = (tile & 255) << 4;
  int n = n0 + lo;
  int h = n >> 6, w = n & 63;

  v8f acc[7];
#pragma unroll
  for (int t = 0; t < 7; ++t)
#pragma unroll
    for (int r = 0; r < 8; ++r) acc[t][r] = 0.f;

  const _Float16* xb = xh + (size_t)b * PADA * CCH + (hi << 4);
  const _Float16* wb = wpk + ((size_t)lane << 4);

#pragma unroll
  for (int kk = 0; kk < 9; ++kk) {
    const int dy = kk / 3, dx = kk % 3;                  // constant-folded
    const _Float16* bp = xb + (size_t)((h + dy) * PADW + (w + dx)) * CCH;
    // speculative prefetch of the next tap's im2col row (gfx1250 global_prefetch)
    if (kk < 8) {
      const int ny = (kk + 1) / 3, nx = (kk + 1) % 3;
      __builtin_prefetch(xb + (size_t)((h + ny) * PADW + (w + nx)) * CCH, 0, 0);
    }
    const _Float16* wk = wb + (size_t)(kk * 8) * 512;
#pragma unroll 2
    for (int cc = 0; cc < 8; ++cc) {
      v16h bm = *(const v16h*)(bp + (cc << 5));
      const _Float16* wp = wk + (size_t)cc * 512;
      v16h am[7];
#pragma unroll
      for (int t = 0; t < 7; ++t)
        am[t] = *(const v16h*)(wp + (size_t)t * 36864);  // t stride 72*512
#pragma unroll
      for (int t = 0; t < 7; ++t)
        acc[t] = __builtin_amdgcn_wmma_f32_16x16x32_f16(false, am[t], false, bm,
                                                        (short)0, acc[t], false, false);
    }
  }
#pragma unroll
  for (int t = 0; t < 7; ++t)
#pragma unroll
    for (int r = 0; r < 8; ++r) {
      int m = t * 16 + r + (hi << 3);
      if (t < 6 || m < 108)
        off[(((size_t)b * 108 + m) << 12) + n] = acc[t][r] + bias[m];
    }
}

// ---- Stage 2: fused bilinear-gather + modulation + grouped GEMM --------------
// one wave per (group, 16-pixel tile): offsets/sigmoid/corner addresses once per
// tap; branchless 4-corner packed-f16 blend (invalid corners get weight 0 and a
// clamped in-bounds address); 4 accumulators reuse each B chunk: 72 WMMAs/wave
__global__ void __launch_bounds__(128)
k_dcn(const _Float16* __restrict__ xh, const _Float16* __restrict__ wpk,
      const float* __restrict__ off, const float* __restrict__ bias,
      float* __restrict__ outn) {
  int lane = threadIdx.x & 31;
  int wv = threadIdx.x >> 5;
  int lo = lane & 15, hi = (lane >> 4) & 1;
  int tile = blockIdx.x * 4 + wv;     // 0..2047
  int b = tile >> 8;
  int n0 = (tile & 255) << 4;
  int g = blockIdx.y;
  int n = n0 + lo;
  int h = n >> 6, w = n & 63;

  v8f acc[4];
#pragma unroll
  for (int t = 0; t < 4; ++t)
#pragma unroll
    for (int r = 0; r < 8; ++r) acc[t][r] = 0.f;

  const float* offb = off + ((size_t)b * 108 << 12);
  const _Float16* xg = xh + (size_t)b * PADA * CCH + g * CG + (hi << 4);
  const _Float16* wdb = wpk + ((size_t)(g << 2) * 18 << 9) + ((size_t)lane << 4);

#pragma unroll
  for (int kk = 0; kk < 9; ++kk) {
    const int dy = kk / 3 - 1, dx = kk % 3 - 1;         // constant-folded
    float oy = offb[(size_t)(g * 18 + kk * 2 + 0) * HWPX + n];
    float ox = offb[(size_t)(g * 18 + kk * 2 + 1) * HWPX + n];
    float ml = offb[(size_t)(72 + g * 9 + kk) * HWPX + n];
    float msk = 1.f / (1.f + __expf(-ml));
    float py = (float)(h + dy) + oy;
    float px = (float)(w + dx) + ox;
    float y0f = floorf(py), x0f = floorf(px);
    float wy1 = py - y0f, wy0 = 1.f - wy1;
    float wx1 = px - x0f, wx0 = 1.f - wx1;
    int y0 = (int)y0f, x0 = (int)x0f;
    int cy[4] = {y0, y0, y0 + 1, y0 + 1};
    int cx[4] = {x0, x0 + 1, x0, x0 + 1};
    float wf[4] = {wy0 * wx0, wy0 * wx1, wy1 * wx0, wy1 * wx1};
    size_t coff[4];
    _Float16 wh[4];
#pragma unroll
    for (int cn = 0; cn < 4; ++cn) {
      int yp = cy[cn] + 1, xp = cx[cn] + 1;             // padded coords
      bool v = ((unsigned)yp < (unsigned)PADW) & ((unsigned)xp < (unsigned)PADW);
      int yi = v ? yp : 0, xi = v ? xp : 0;
      coff[cn] = (size_t)(yi * PADW + xi) * CCH;
      wh[cn] = (_Float16)(v ? wf[cn] * msk : 0.f);
    }

#pragma unroll
    for (int cc = 0; cc < 2; ++cc) {
      const _Float16* base = xg + (cc << 5);
      v16h bm = (*(const v16h*)(base + coff[0])) * wh[0];
      bm += (*(const v16h*)(base + coff[1])) * wh[1];
      bm += (*(const v16h*)(base + coff[2])) * wh[2];
      bm += (*(const v16h*)(base + coff[3])) * wh[3];

      const _Float16* wp = wdb + (size_t)(kk * 2 + cc) * 512;
      v16h am[4];
#pragma unroll
      for (int t = 0; t < 4; ++t)
        am[t] = *(const v16h*)(wp + (size_t)t * 9216);  // t stride 18*512
#pragma unroll
      for (int t = 0; t < 4; ++t)
        acc[t] = __builtin_amdgcn_wmma_f32_16x16x32_f16(false, am[t], false, bm,
                                                        (short)0, acc[t], false, false);
    }
  }
  size_t po = (((size_t)b << 12) + n) * CCH + g * CG;
#pragma unroll
  for (int t = 0; t < 4; ++t)
#pragma unroll
    for (int r = 0; r < 8; ++r) {
      int oc = t * 16 + r + (hi << 3);
      outn[po + oc] = acc[t][r] + bias[g * CG + oc];
    }
}

// ---- Stage 3: LayerNorm over C + sigmoid, in-place on NHWC buffer ------------
__global__ void k_ln(float* __restrict__ attn, const float* __restrict__ gamma,
                     const float* __restrict__ beta) {
  int wave = threadIdx.x >> 5;
  int lane = threadIdx.x & 31;
  int p = blockIdx.x * 8 + wave;      // 32768 pixels total
  float* row = attn + (size_t)p * CCH;
  float v[8];
#pragma unroll
  for (int i = 0; i < 8; ++i) v[i] = row[lane * 8 + i];
  float s = 0.f;
#pragma unroll
  for (int i = 0; i < 8; ++i) s += v[i];
#pragma unroll
  for (int o = 16; o > 0; o >>= 1) s += __shfl_xor(s, o, 32);
  float mu = s * (1.f / 256.f);
  float q = 0.f;
#pragma unroll
  for (int i = 0; i < 8; ++i) { float d = v[i] - mu; q += d * d; }
#pragma unroll
  for (int o = 16; o > 0; o >>= 1) q += __shfl_xor(q, o, 32);
  float rstd = rsqrtf(q * (1.f / 256.f) + 1e-5f);
#pragma unroll
  for (int i = 0; i < 8; ++i) {
    int c = lane * 8 + i;
    float z = (v[i] - mu) * rstd * gamma[c] + beta[c];
    row[c] = 1.f / (1.f + __expf(-z));
  }
}

// ---- Stage 4: LDS-tiled transpose-multiply: out(NCHW) = x(NCHW) * attn(NHWC) -
__global__ void k_txm(const float* __restrict__ x, const float* __restrict__ attn,
                      float* __restrict__ out) {
  __shared__ float tile[64][65];
  int bx = blockIdx.x;                // b*64 + pixel-tile
  int b = bx >> 6;
  int p0 = (bx & 63) << 6;
  int c0 = blockIdx.y << 6;
  int tj = threadIdx.x & 63;
  int ti = threadIdx.x >> 6;          // 4 rows per pass
#pragma unroll
  for (int r = 0; r < 16; ++r) {
    int pp = ti + r * 4;
    tile[pp][tj] = attn[(((size_t)b << 12) + p0 + pp) * CCH + c0 + tj];
  }
  __syncthreads();
#pragma unroll
  for (int r = 0; r < 16; ++r) {
    int cc = ti + r * 4;
    size_t idx = (((size_t)b * CCH + c0 + cc) << 12) + p0 + tj;
    out[idx] = x[idx] * tile[tj][cc];
  }
}

extern "C" void kernel_launch(void* const* d_in, const int* in_sizes, int n_in,
                              void* d_out, int out_size, void* d_ws, size_t ws_size,
                              hipStream_t stream) {
  const float* x     = (const float*)d_in[0];
  const float* w_off = (const float*)d_in[1];
  const float* b_off = (const float*)d_in[2];
  const float* w_dcn = (const float*)d_in[3];
  const float* b_dcn = (const float*)d_in[4];
  const float* gamma = (const float*)d_in[5];
  const float* beta  = (const float*)d_in[6];
  float* out = (float*)d_out;

  // workspace layout (bytes), ~66.4 MB total
  char* ws = (char*)d_ws;
  _Float16* xh      = (_Float16*)(ws);              // 17,842,176 (padded NHWC)
  _Float16* woff_pk = (_Float16*)(ws + 17842176);   //    516,096
  _Float16* wdcn_pk = (_Float16*)(ws + 18358272);   //    294,912
  float*    off_ws  = (float*)(ws + 18653184);      // 14,155,776
  float*    attn    = (float*)(ws + 32808960);      // 33,554,432

  k_convert_x<<<dim3(PADA, BATCH), dim3(256), 0, stream>>>(x, xh);
  k_pack_woff<<<dim3(72, 7), dim3(512), 0, stream>>>(w_off, woff_pk);
  k_pack_wdcn<<<dim3(18, 16), dim3(512), 0, stream>>>(w_dcn, wdcn_pk);
  k_conv_off<<<dim3(512), dim3(128), 0, stream>>>(xh, woff_pk, b_off, off_ws);
  k_dcn<<<dim3(512, 4), dim3(128), 0, stream>>>(xh, wdcn_pk, off_ws, b_dcn, attn);
  k_ln<<<dim3(4096), dim3(256), 0, stream>>>(attn, gamma, beta);
  k_txm<<<dim3(512, 4), dim3(256), 0, stream>>>(x, attn, out);
}
